// VQQuantizer_53266184405016
// MI455X (gfx1250) — matrix operand order repaired and checked by reading, not compiled, not run on hardware
//
#include <hip/hip_runtime.h>

#define N_ROWS 32768
#define DIM    512
#define K_CB   1024

typedef __attribute__((ext_vector_type(16))) _Float16 v16h;
typedef __attribute__((ext_vector_type(8)))  float    v8f;

union Frag16 { v16h v; uint4 u[2]; };
union H4     { _Float16 h[4]; uint2 u; };

// ---------------------------------------------------------------------------
// Kernel 1/2: L2-normalize rows of a [nrows, 512] f32 matrix.
// One wave (32 lanes) per row; each lane handles 16 floats (4 x float4).
// Writes f16 copy (for WMMA) and optionally an f32 copy.
// ---------------------------------------------------------------------------
__global__ void vq_normalize_rows(const float* __restrict__ src,
                                  _Float16* __restrict__ dstH,
                                  float* __restrict__ dstF,
                                  int nrows) {
    const int lane = threadIdx.x & 31;
    const int wave = threadIdx.x >> 5;
    const int row  = blockIdx.x * 8 + wave;
    if (row >= nrows) return;

    const float4* s = (const float4*)(src + (size_t)row * DIM);
    float4 x[4];
    float ss = 0.0f;
#pragma unroll
    for (int i = 0; i < 4; ++i) {
        x[i] = s[lane + 32 * i];
        ss += x[i].x * x[i].x + x[i].y * x[i].y + x[i].z * x[i].z + x[i].w * x[i].w;
    }
#pragma unroll
    for (int off = 1; off < 32; off <<= 1) ss += __shfl_xor(ss, off, 32);
    const float inv = 1.0f / fmaxf(sqrtf(ss), 1e-6f);

#pragma unroll
    for (int i = 0; i < 4; ++i) {
        float4 y;
        y.x = x[i].x * inv; y.y = x[i].y * inv;
        y.z = x[i].z * inv; y.w = x[i].w * inv;
        if (dstF) ((float4*)(dstF + (size_t)row * DIM))[lane + 32 * i] = y;
        H4 p;
        p.h[0] = (_Float16)y.x; p.h[1] = (_Float16)y.y;
        p.h[2] = (_Float16)y.z; p.h[3] = (_Float16)y.w;
        ((uint2*)(dstH + (size_t)row * DIM))[lane + 32 * i] = p.u;
    }
}

__global__ void vq_zero_loss(float* __restrict__ loss) { *loss = 0.0f; }

// ---------------------------------------------------------------------------
// Kernel 3: fused GEMM (h_norm @ cb^T via v_wmma_f32_16x16x32_f16) + argmax.
// Block = 256 threads = 8 waves; each wave owns a 16-row tile (block: 128 rows).
// The wave's full A tile (16 rows x 512 halves = 128 VGPRs) is held in
// registers for the whole kernel (launch_bounds(256,1) provides the budget).
// B tiles (16 codebook entries x 512 halves = 16 KB) are double-buffered in
// LDS and shared by all 8 waves; next tile's global loads are issued before
// the 16-WMMA compute block so they overlap with the matrix pipe.
// argmin(dist) == argmax(dot) because both operands are L2-normalized.
// ---------------------------------------------------------------------------
__global__ __launch_bounds__(256, 1)
void vq_wmma_argmax(const _Float16* __restrict__ hA,   // [N_ROWS, DIM] f16
                    const _Float16* __restrict__ cbB,  // [K_CB, DIM]  f16
                    int* __restrict__ indices) {       // [N_ROWS]
    __shared__ _Float16 Bt[2 * 16 * DIM];   // 2 x 16 KB double buffer

    const int lane    = threadIdx.x & 31;
    const int wave    = threadIdx.x >> 5;
    const int half    = lane >> 4;      // 0: lanes 0-15, 1: lanes 16-31
    const int lid     = lane & 15;
    const int rowBase = blockIdx.x * 128 + wave * 16;

    // Preload this wave's entire A tile into VGPRs.
    // 16-bit A 16x32 layout: lane holds row (rowBase+lid); first 8 halves at
    // K = kk*32 + half*8, second 8 halves at K = kk*32 + 16 + half*8.
    const _Float16* aptr = hA + (size_t)(rowBase + lid) * DIM + half * 8;
    Frag16 a[16];
#pragma unroll
    for (int kk = 0; kk < 16; ++kk) {
        a[kk].u[0] = *(const uint4*)(aptr + kk * 32);
        a[kk].u[1] = *(const uint4*)(aptr + kk * 32 + 16);
    }

    // Stage B tile 0 into buffer 0 (1024 uint4, 4 per thread).
    {
        const uint4* s = (const uint4*)cbB;
        uint4* d = (uint4*)Bt;
#pragma unroll
        for (int i = 0; i < 4; ++i) d[threadIdx.x + 256 * i] = s[threadIdx.x + 256 * i];
    }
    __syncthreads();

    float best[8];
    int   bestIdx[8];
#pragma unroll
    for (int r = 0; r < 8; ++r) { best[r] = -3.0e38f; bestIdx[r] = 0; }

    for (int j = 0; j < 64; ++j) {
        const int buf = j & 1;

        // Issue next tile's global loads early (overlap with WMMA below).
        uint4 stage[4];
        if (j < 63) {
            const uint4* s = (const uint4*)(cbB + (size_t)(j + 1) * 16 * DIM);
#pragma unroll
            for (int i = 0; i < 4; ++i) stage[i] = s[threadIdx.x + 256 * i];
        }

        // B fragment base (LDS): lane holds entry lid of the tile; first 8
        // halves at K = kk*32 + half*8, second 8 at K = kk*32 + 16 + half*8.
        const _Float16* bptr = Bt + buf * (16 * DIM) + (size_t)lid * DIM + half * 8;

        v8f acc = {};
#pragma unroll
        for (int kk = 0; kk < 16; ++kk) {
            Frag16 b;
            b.u[0] = *(const uint4*)(bptr + kk * 32);
            b.u[1] = *(const uint4*)(bptr + kk * 32 + 16);
            acc = __builtin_amdgcn_wmma_f32_16x16x32_f16(
                false, a[kk].v, false, b.v, (short)0, acc, false, false);
        }

        // Commit next tile to the alternate buffer (was last read in j-1).
        if (j < 63) {
            uint4* d = (uint4*)(Bt + (1 - buf) * (16 * DIM));
#pragma unroll
            for (int i = 0; i < 4; ++i) d[threadIdx.x + 256 * i] = stage[i];
        }
        __syncthreads();

        // Running per-lane argmax: C layout puts column (j*16 + lid) in every
        // acc slot of this lane; row = rowBase + r + half*8.
        const int col = j * 16 + lid;
#pragma unroll
        for (int r = 0; r < 8; ++r) {
            if (acc[r] > best[r]) { best[r] = acc[r]; bestIdx[r] = col; }
        }
    }

    // Cross-lane argmax within each 16-lane half. Tie -> lowest index,
    // matching argmin-first-occurrence semantics.
#pragma unroll
    for (int r = 0; r < 8; ++r) {
        float v = best[r];
        int   ix = bestIdx[r];
#pragma unroll
        for (int off = 1; off < 16; off <<= 1) {
            const float ov = __shfl_xor(v, off, 32);
            const int   oi = __shfl_xor(ix, off, 32);
            if (ov > v || (ov == v && oi < ix)) { v = ov; ix = oi; }
        }
        if (lid == 0) indices[rowBase + r + half * 8] = ix;
    }
}

// ---------------------------------------------------------------------------
// Kernel 4: per-row finalize. Recomputes h_norm scale, gathers cb_norm[idx]
// into c_tilde/c_hard/c_quantized (identical forward values), writes one-hot
// q row, accumulates loss = (1+BETA) * mean(diff^2).
// One block (256 threads) per row.
// ---------------------------------------------------------------------------
__global__ void vq_finalize(const float* __restrict__ h,     // raw h [N_ROWS, DIM]
                            const float* __restrict__ cbn,   // cb_norm [K_CB, DIM]
                            const int* __restrict__ idxArr,
                            float* __restrict__ q,
                            float* __restrict__ c_tilde,
                            float* __restrict__ c_hard,
                            float* __restrict__ c_quant,
                            float* __restrict__ loss,
                            float* __restrict__ idxOut) {
    __shared__ float red[8];
    const size_t n    = blockIdx.x;
    const int    t    = threadIdx.x;
    const int    lane = t & 31;
    const int    wave = t >> 5;
    const int    idx  = idxArr[n];

    // recompute row norm of h
    const float2 hv = ((const float2*)(h + n * DIM))[t];
    float ssh = hv.x * hv.x + hv.y * hv.y;
#pragma unroll
    for (int off = 1; off < 32; off <<= 1) ssh += __shfl_xor(ssh, off, 32);
    if (lane == 0) red[wave] = ssh;
    __syncthreads();
    float tot = 0.0f;
#pragma unroll
    for (int i = 0; i < 8; ++i) tot += red[i];
    const float inv = 1.0f / fmaxf(sqrtf(tot), 1e-6f);

    // gather code vector; forward c_tilde == c_hard == c_quantized
    const float2 c = ((const float2*)(cbn + (size_t)idx * DIM))[t];
    ((float2*)(c_tilde + n * DIM))[t] = c;
    ((float2*)(c_hard  + n * DIM))[t] = c;
    ((float2*)(c_quant + n * DIM))[t] = c;

    const float dx = hv.x * inv - c.x;
    const float dy = hv.y * inv - c.y;
    float ss = dx * dx + dy * dy;

    // one-hot q row: 1024 cols, 4 per thread
    const int cbase = t * 4;
    float4 qv;
    qv.x = (idx == cbase + 0) ? 1.0f : 0.0f;
    qv.y = (idx == cbase + 1) ? 1.0f : 0.0f;
    qv.z = (idx == cbase + 2) ? 1.0f : 0.0f;
    qv.w = (idx == cbase + 3) ? 1.0f : 0.0f;
    ((float4*)(q + n * (size_t)K_CB))[t] = qv;

    // loss partial: (1 + 0.25) / (N*D)
#pragma unroll
    for (int off = 1; off < 32; off <<= 1) ss += __shfl_xor(ss, off, 32);
    __syncthreads();
    if (lane == 0) red[wave] = ss;
    __syncthreads();
    if (t == 0) {
        float s = 0.0f;
#pragma unroll
        for (int i = 0; i < 8; ++i) s += red[i];
        atomicAdd(loss, s * (1.25f / ((float)N_ROWS * (float)DIM)));
        idxOut[n] = (float)idx;
    }
}

// ---------------------------------------------------------------------------
extern "C" void kernel_launch(void* const* d_in, const int* in_sizes, int n_in,
                              void* d_out, int out_size, void* d_ws, size_t ws_size,
                              hipStream_t stream) {
    const float* h  = (const float*)d_in[0];   // [32,1024,512] f32
    const float* cb = (const float*)d_in[1];   // [1024,512] f32
    (void)in_sizes; (void)n_in; (void)out_size; (void)ws_size;

    // Output layout (concatenated, return order):
    float* out      = (float*)d_out;
    float* q        = out;                                   // 32768*1024
    float* c_tilde  = q + (size_t)N_ROWS * K_CB;             // 32768*512
    float* c_hard   = c_tilde + (size_t)N_ROWS * DIM;
    float* c_quant  = c_hard + (size_t)N_ROWS * DIM;
    float* loss     = c_quant + (size_t)N_ROWS * DIM;        // scalar
    float* idxOut   = loss + 1;                              // 32768

    // Workspace layout
    char* ws = (char*)d_ws;
    _Float16* hA   = (_Float16*)ws;                                   // 33,554,432 B
    _Float16* cbB  = (_Float16*)(ws + 33554432ull);                   //  1,048,576 B
    float*    cbn  = (float*)(ws + 33554432ull + 1048576ull);         //  2,097,152 B
    int*      idxW = (int*)(ws + 33554432ull + 1048576ull + 2097152ull); // 131,072 B

    // 1) normalize h -> f16 (f32 copy not needed; finalize recomputes scale)
    vq_normalize_rows<<<N_ROWS / 8, 256, 0, stream>>>(h, hA, nullptr, N_ROWS);
    // 2) normalize codebook -> f16 (WMMA B) + f32 (gather source)
    vq_normalize_rows<<<K_CB / 8, 256, 0, stream>>>(cb, cbB, cbn, K_CB);
    vq_zero_loss<<<1, 1, 0, stream>>>(loss);
    // 3) WMMA GEMM + fused argmax (argmin of distance == argmax of dot)
    vq_wmma_argmax<<<N_ROWS / 128, 256, 0, stream>>>(hA, cbB, idxW);
    // 4) gather outputs, one-hot q, loss
    vq_finalize<<<N_ROWS, 256, 0, stream>>>(h, cbn, idxW, q, c_tilde, c_hard,
                                            c_quant, loss, idxOut);
}
